// MultiheadAttention_7808250544759
// MI455X (gfx1250) — compile-verified, matrix operand order checked by reading, not run on hardware
//
#include <hip/hip_runtime.h>
#include <cstdint>
#include <cstddef>

// ---------------------------------------------------------------------------
// Types
// ---------------------------------------------------------------------------
typedef __bf16 bf16;
typedef __attribute__((ext_vector_type(16))) __bf16 bf16x16;
typedef __attribute__((ext_vector_type(8)))  __bf16 bf16x8;
typedef __attribute__((ext_vector_type(4)))  __bf16 bf16x4;
typedef __attribute__((ext_vector_type(8)))  float  f32x8;
typedef __attribute__((ext_vector_type(4)))  float  f32x4;

#define DEV static __device__ __forceinline__

// Problem constants
constexpr int Bb = 2;      // batch
constexpr int Ss = 2048;   // sequence
constexpr int Dd = 1024;   // model dim
constexpr int Hh = 16;     // heads
constexpr int DK = 64;     // head dim
constexpr int HS = Ss * DK;  // per-head contiguous chunk (faithful reshape quirk)

// ---------------------------------------------------------------------------
// WMMA helpers
// ---------------------------------------------------------------------------
// 16-bit A-matrix 16x32 fragment (also B^T-row fragment): lane l in [0,16)
// holds row (l), K 0..7 / 16..23; lane l+16 holds row l, K 8..15 / 24..31.
DEV bf16x16 frag_ld(const bf16* p0, int stride, int lane) {
  const bf16* p = p0 + (lane & 15) * stride + ((lane >> 4) << 3);
  bf16x8 lo = *(const bf16x8*)(p);
  bf16x8 hi = *(const bf16x8*)(p + 16);
  bf16x16 f;
#pragma unroll
  for (int i = 0; i < 8; ++i) { f[i] = lo[i]; f[i + 8] = hi[i]; }
  return f;
}

DEV f32x8 wmma_bf16(bf16x16 a, bf16x16 b, f32x8 c) {
  // 8 args: (neg_a, A, neg_b, B, c_mod, C, reuse_a, reuse_b)
  return __builtin_amdgcn_wmma_f32_16x16x32_bf16(
      false, a, false, b, (short)0, c, false, false);
}

// Row reductions inside 16-lane halves (matches the C-matrix lane layout:
// reg r holds row r in lanes 0-15 and row r+8 in lanes 16-31).
DEV float red_max16(float v) {
#pragma unroll
  for (int m = 1; m < 16; m <<= 1) v = fmaxf(v, __shfl_xor(v, m, 32));
  return v;
}
DEV float red_sum16(float v) {
#pragma unroll
  for (int m = 1; m < 16; m <<= 1) v += __shfl_xor(v, m, 32);
  return v;
}

// ---------------------------------------------------------------------------
// GEMM:  C[M,N] = A[M,K] @ W[K,N] + bias     (A: f32 or bf16, C: bf16 or f32)
// Block tile 128x128, BK=64, 256 threads = 8 waves in a 2x4 grid;
// each wave computes 64x32 = 4x2 WMMA tiles.
// ---------------------------------------------------------------------------
template <bool A_BF16, bool OUT_BF16>
__global__ __launch_bounds__(256) void gemm_bias_kernel(
    const void* __restrict__ Aptr, const float* __restrict__ W,
    const float* __restrict__ bias, void* __restrict__ Cptr,
    int M, int N, int K) {
  constexpr int BM = 128, BN = 128, BK = 64, LDT = 72;
  __shared__ bf16 As[BM][LDT];   // A tile, row-major [m][k]
  __shared__ bf16 Bs[BN][LDT];   // W tile transposed: [n][k]

  const int tid  = threadIdx.x;
  const int lane = tid & 31;
  const int wave = tid >> 5;
  const int wm = wave >> 2;      // 0..1  (64-row slab)
  const int wn = wave & 3;       // 0..3  (32-col slab)
  const int mBase = blockIdx.y * BM;
  const int nBase = blockIdx.x * BN;

  f32x8 acc[4][2];
#pragma unroll
  for (int i = 0; i < 4; ++i)
#pragma unroll
    for (int j = 0; j < 2; ++j) acc[i][j] = f32x8{};

  for (int k0 = 0; k0 < K; k0 += BK) {
    // ---- stage A tile (convert f32 -> bf16 if needed) ----
    {
      const int row = tid >> 1;
      const int off = (tid & 1) * 32;
      if constexpr (A_BF16) {
        const bf16* A = (const bf16*)Aptr + (size_t)(mBase + row) * K + k0 + off;
#pragma unroll
        for (int i = 0; i < 4; ++i)
          *(bf16x8*)(&As[row][off + i * 8]) = *(const bf16x8*)(A + i * 8);
      } else {
        const float* A = (const float*)Aptr + (size_t)(mBase + row) * K + k0 + off;
#pragma unroll
        for (int i = 0; i < 8; ++i) {
          f32x4 v = *(const f32x4*)(A + i * 4);
          bf16x4 o;
#pragma unroll
          for (int j = 0; j < 4; ++j) o[j] = (bf16)v[j];
          *(bf16x4*)(&As[row][off + i * 4]) = o;
        }
      }
    }
    // ---- stage W tile, transposed into Bs[n][k] ----
    {
      const int kr   = tid >> 2;         // 0..63
      const int nOff = (tid & 3) * 32;   // 0..96
      const float* Wp = W + (size_t)(k0 + kr) * N + nBase + nOff;
#pragma unroll
      for (int i = 0; i < 8; ++i) {
        f32x4 v = *(const f32x4*)(Wp + i * 4);
#pragma unroll
        for (int j = 0; j < 4; ++j) Bs[nOff + i * 4 + j][kr] = (bf16)v[j];
      }
    }
    // prefetch next k-tiles into cache (global_prefetch)
    if (k0 + BK < K) {
      const char* an = (const char*)Aptr +
          ((size_t)(mBase + (tid >> 1)) * K + k0 + BK) * (A_BF16 ? 2 : 4);
      __builtin_prefetch(an, 0, 1);
      __builtin_prefetch(W + (size_t)(k0 + BK + (tid >> 2)) * N + nBase, 0, 1);
    }
    __syncthreads();

    // ---- compute: 2 k-steps of 32, 4x2 WMMA tiles each ----
#pragma unroll
    for (int kk = 0; kk < BK; kk += 32) {
      bf16x16 af[4], bfr[2];
#pragma unroll
      for (int i = 0; i < 4; ++i)
        af[i] = frag_ld(&As[wm * 64 + i * 16][kk], LDT, lane);
#pragma unroll
      for (int j = 0; j < 2; ++j)
        bfr[j] = frag_ld(&Bs[wn * 32 + j * 16][kk], LDT, lane);
#pragma unroll
      for (int i = 0; i < 4; ++i)
#pragma unroll
        for (int j = 0; j < 2; ++j)
          acc[i][j] = wmma_bf16(af[i], bfr[j], acc[i][j]);
    }
    __syncthreads();
  }

  // ---- epilogue: + bias, store (C layout: reg p -> row p (+8 for hi half)) --
  const int rAdd = 8 * (lane >> 4);
  const int cAdd = lane & 15;
#pragma unroll
  for (int i = 0; i < 4; ++i)
#pragma unroll
    for (int j = 0; j < 2; ++j) {
      const int col = nBase + wn * 32 + j * 16 + cAdd;
      const float bv = bias[col];
#pragma unroll
      for (int p = 0; p < 8; ++p) {
        const int row = mBase + wm * 64 + i * 16 + p + rAdd;
        const float val = acc[i][j][p] + bv;
        if constexpr (OUT_BF16)
          ((bf16*)Cptr)[(size_t)row * N + col] = (bf16)val;
        else
          ((float*)Cptr)[(size_t)row * N + col] = val;
      }
    }
}

// ---------------------------------------------------------------------------
// Flash attention over per-head [2048,64] contiguous bf16 Q/K/V chunks.
// Grid: (S/128, B*H). Block: 256 threads = 8 waves; wave owns 16 query rows.
// Faithful quirks honored: multiplicative mask before softmax, and
// post-softmax /sqrt(dk) folded into the final normalization (O /= l*8).
// ---------------------------------------------------------------------------
__global__ __launch_bounds__(256) void attn_kernel(
    const bf16* __restrict__ Qb, const bf16* __restrict__ Kb,
    const bf16* __restrict__ Vb, const float* __restrict__ mask,
    bf16* __restrict__ Ob) {
  constexpr int LK = 72;    // Ks row stride (64 feat + pad)
  constexpr int LP = 136;   // Vt/Ps row stride (128 keys + pad)
  __shared__ bf16 Ks[128][LK];      // K block rows   [key][feat]
  __shared__ bf16 Vt[64][LP];       // V block transposed [feat][key]
  __shared__ bf16 Ps[8][16][LP];    // per-wave P scratch [qrow][key]

  const int tid  = threadIdx.x;
  const int lane = tid & 31;
  const int wave = tid >> 5;
  const int bh = blockIdx.y;
  const int b  = bh >> 4;
  const int h  = bh & 15;
  const bf16* Qh = Qb + (size_t)b * Ss * Dd + (size_t)h * HS;
  const bf16* Kh = Kb + (size_t)b * Ss * Dd + (size_t)h * HS;
  const bf16* Vh = Vb + (size_t)b * Ss * Dd + (size_t)h * HS;
  const int qw = blockIdx.x * 128 + wave * 16;   // this wave's first q row

  // Q fragments (16 rows x 64 feat) loaded once, straight from global
  bf16x16 qf[2];
#pragma unroll
  for (int c = 0; c < 2; ++c)
    qf[c] = frag_ld(Qh + (size_t)qw * DK + c * 32, DK, lane);

  float m_[8], l_[8];
  f32x8 o_[4];
#pragma unroll
  for (int p = 0; p < 8; ++p) { m_[p] = -1e30f; l_[p] = 0.0f; }
#pragma unroll
  for (int t = 0; t < 4; ++t) o_[t] = f32x8{};

  const int rAdd = 8 * (lane >> 4);
  const int cAdd = lane & 15;

  for (int j = 0; j < Ss / 128; ++j) {
    const int kb = j * 128;
    // ---- stage K rows and V^T into LDS ----
    {
      const int row = tid >> 1;
      const int off = (tid & 1) * 32;
      const bf16* ks = Kh + (size_t)(kb + row) * DK + off;
      const bf16* vs = Vh + (size_t)(kb + row) * DK + off;
#pragma unroll
      for (int i = 0; i < 4; ++i)
        *(bf16x8*)(&Ks[row][off + i * 8]) = *(const bf16x8*)(ks + i * 8);
#pragma unroll
      for (int i = 0; i < 4; ++i) {
        bf16x8 v = *(const bf16x8*)(vs + i * 8);
#pragma unroll
        for (int e = 0; e < 8; ++e) Vt[off + i * 8 + e][row] = v[e];
      }
    }
    __syncthreads();

    // ---- S = Q @ K^T  (16 x 128 per wave) ----
    f32x8 sc[8];
#pragma unroll
    for (int nt = 0; nt < 8; ++nt) sc[nt] = f32x8{};
#pragma unroll
    for (int nt = 0; nt < 8; ++nt)
#pragma unroll
      for (int c = 0; c < 2; ++c) {
        bf16x16 kf = frag_ld(&Ks[nt * 16][c * 32], LK, lane);
        sc[nt] = wmma_bf16(qf[c], kf, sc[nt]);
      }

    // ---- multiplicative mask (faithful) ----
    {
      const float* mb = mask + ((size_t)b * Ss + (qw + rAdd)) * Ss + kb + cAdd;
#pragma unroll
      for (int p = 0; p < 8; ++p) {
        const float* mp = mb + (size_t)p * Ss;
#pragma unroll
        for (int nt = 0; nt < 8; ++nt) sc[nt][p] *= mp[nt * 16];
      }
    }

    // ---- online softmax: row max / exp / row sum ----
#pragma unroll
    for (int p = 0; p < 8; ++p) {
      float mx = sc[0][p];
#pragma unroll
      for (int nt = 1; nt < 8; ++nt) mx = fmaxf(mx, sc[nt][p]);
      mx = red_max16(mx);
      const float mn = fmaxf(m_[p], mx);
      const float alpha = __expf(m_[p] - mn);
      m_[p] = mn;
      float rs = 0.0f;
#pragma unroll
      for (int nt = 0; nt < 8; ++nt) {
        const float e = __expf(sc[nt][p] - mn);
        sc[nt][p] = e;
        rs += e;
      }
      rs = red_sum16(rs);
      l_[p] = l_[p] * alpha + rs;
#pragma unroll
      for (int t = 0; t < 4; ++t) o_[t][p] *= alpha;
    }

    // ---- P -> LDS (bf16, A-fragment friendly layout) ----
#pragma unroll
    for (int nt = 0; nt < 8; ++nt)
#pragma unroll
      for (int p = 0; p < 8; ++p)
        Ps[wave][p + rAdd][nt * 16 + cAdd] = (bf16)sc[nt][p];
    // same-wave DS RAW: wait for the stores before the fragment reloads
    asm volatile("s_wait_dscnt 0" ::: "memory");

    // ---- O += P @ V ----
#pragma unroll
    for (int c = 0; c < 4; ++c) {
      bf16x16 pf = frag_ld(&Ps[wave][0][c * 32], LP, lane);
#pragma unroll
      for (int t = 0; t < 4; ++t) {
        bf16x16 vf = frag_ld(&Vt[t * 16][c * 32], LP, lane);
        o_[t] = wmma_bf16(pf, vf, o_[t]);
      }
    }
    __syncthreads();
  }

  // ---- normalize (softmax denom * post-softmax /sqrt(64)) and write back ----
  // out[b, q, h*64 + dk]  (this is transpose(0,2,1,3).reshape under the quirk)
#pragma unroll
  for (int p = 0; p < 8; ++p) {
    const float inv = 1.0f / (l_[p] * 8.0f);
    const int q = qw + p + rAdd;
#pragma unroll
    for (int t = 0; t < 4; ++t) {
      const float val = o_[t][p] * inv;
      Ob[(size_t)b * Ss * Dd + (size_t)q * Dd + h * 64 + t * 16 + cAdd] =
          (bf16)val;
    }
  }
}

// ---------------------------------------------------------------------------
// Launcher
// ---------------------------------------------------------------------------
extern "C" void kernel_launch(void* const* d_in, const int* in_sizes, int n_in,
                              void* d_out, int out_size, void* d_ws,
                              size_t ws_size, hipStream_t stream) {
  const float* target = (const float*)d_in[0];
  const float* source = (const float*)d_in[1];
  const float* memory = (const float*)d_in[2];
  const float* mask   = (const float*)d_in[3];
  const float* WQ = (const float*)d_in[4];
  const float* bQ = (const float*)d_in[5];
  const float* WK = (const float*)d_in[6];
  const float* bK = (const float*)d_in[7];
  const float* WV = (const float*)d_in[8];
  const float* bV = (const float*)d_in[9];
  const float* Wp = (const float*)d_in[10];
  const float* bp = (const float*)d_in[11];

  const int M = Bb * Ss;            // 4096
  const int N = Dd, K = Dd;         // 1024
  const size_t elems = (size_t)M * Dd;       // per-buffer bf16 elements
  bf16* Qb = (bf16*)d_ws;
  bf16* Kb = Qb + elems;
  bf16* Vb = Kb + elems;
  bf16* Ob = Vb + elems;            // needs 4 * 8MB = 32MB of workspace

  dim3 blk(256);
  dim3 ggrid(N / 128, M / 128);     // 8 x 32

  // Q/K/V projections: f32 in -> bf16 out, WMMA bf16 accumulation in f32
  gemm_bias_kernel<false, true><<<ggrid, blk, 0, stream>>>(target, WQ, bQ, Qb, M, N, K);
  gemm_bias_kernel<false, true><<<ggrid, blk, 0, stream>>>(source, WK, bK, Kb, M, N, K);
  gemm_bias_kernel<false, true><<<ggrid, blk, 0, stream>>>(memory, WV, bV, Vb, M, N, K);

  // Flash attention per (batch, head)
  attn_kernel<<<dim3(Ss / 128, Bb * Hh), blk, 0, stream>>>(Qb, Kb, Vb, mask, Ob);

  // Output projection: bf16 in -> f32 out
  gemm_bias_kernel<true, false><<<ggrid, blk, 0, stream>>>(Ob, Wp, bp, (float*)d_out, M, N, K);
}
